// SelfAttention_v1_68736656605705
// MI455X (gfx1250) — matrix operand order repaired and checked by reading, not compiled, not run on hardware
//
#include <hip/hip_runtime.h>

// ---------------------------------------------------------------------------
// out = (X WQ)(X WK)^T (X WV)  ==  X * ( WQ * ( WK^T * (X^T X) * WV ) )
// (no softmax in the reference -> pure matmul chain -> reassociate; ~40 GFLOP
//  instead of ~326 GFLOP, no 8192x8192 score matrix)
//
// Pre-pass: convert everything to bf16 ONCE, in the exact layout each GEMM
// wants (A: MxK row-major, B: pre-transposed NxK). Intermediates are stored
// transposed bf16 straight out of the WMMA accumulators. The GEMM inner loop
// then stages tiles with GLOBAL_LOAD_ASYNC_TO_LDS_B128 (ASYNCcnt DMA, no
// VGPR round-trip, no conversion), double-buffered.
// ---------------------------------------------------------------------------

typedef __attribute__((ext_vector_type(16))) __bf16 v16bf;
typedef __attribute__((ext_vector_type(8)))  __bf16 v8bf;
typedef __attribute__((ext_vector_type(4)))  __bf16 v4bf;
typedef __attribute__((ext_vector_type(8)))  float  v8f;
typedef int v4i __attribute__((vector_size(4 * sizeof(int))));

#define AS1 __attribute__((address_space(1)))
#define AS3 __attribute__((address_space(3)))

#define BM 128
#define BN 128
#define BK 64
#define SA 72   // LDS row stride in halves (64 + 8 pad)
#define SB 72

__device__ __forceinline__ __bf16 f2bf(float x) {
    unsigned u = __builtin_bit_cast(unsigned, x);
    unsigned r = u + 0x7FFFu + ((u >> 16) & 1u);
    unsigned short h = (unsigned short)(r >> 16);
    return __builtin_bit_cast(__bf16, h);
}

__device__ __forceinline__ v16bf cat8(v8bf lo, v8bf hi) {
    return __builtin_shufflevector(lo, hi,
        0,1,2,3,4,5,6,7,8,9,10,11,12,13,14,15);
}

// 16-byte global -> LDS copy. Prefer the CDNA5 async DMA path (ASYNCcnt).
__device__ __forceinline__ void async_copy16(const __bf16* g, __bf16* l) {
#if __has_builtin(__builtin_amdgcn_global_load_async_to_lds_b128)
    __builtin_amdgcn_global_load_async_to_lds_b128(
        (AS1 v4i*)const_cast<__bf16*>(g), (AS3 v4i*)l, 0, 0);
#else
    *(uint4*)l = *(const uint4*)g;
#endif
}

__device__ __forceinline__ void wait_async_all() {
#if __has_builtin(__builtin_amdgcn_s_wait_asynccnt)
    __builtin_amdgcn_s_wait_asynccnt(0);
#else
    asm volatile("s_wait_asynccnt 0x0" ::: "memory");
#endif
}

// ---------------------------------------------------------------------------
// fp32 -> bf16 elementwise convert (n multiple of 2048)
__global__ __launch_bounds__(256)
void cvt_f32_bf16(const float* __restrict__ s, __bf16* __restrict__ d) {
    const size_t i = ((size_t)blockIdx.x * 256 + threadIdx.x) * 8;
    const float4 a = *(const float4*)(s + i);
    const float4 b = *(const float4*)(s + i + 4);
    v8bf o;
    o[0] = f2bf(a.x); o[1] = f2bf(a.y); o[2] = f2bf(a.z); o[3] = f2bf(a.w);
    o[4] = f2bf(b.x); o[5] = f2bf(b.y); o[6] = f2bf(b.z); o[7] = f2bf(b.w);
    *(v8bf*)(d + i) = o;
}

// fp32 (RxC) -> bf16 transposed (CxR). R, C multiples of 64.
__global__ __launch_bounds__(256)
void tr_f32_bf16(const float* __restrict__ s, __bf16* __restrict__ d,
                 int R, int C) {
    __shared__ __bf16 t[64 * 72];
    const int tid = threadIdx.x;
    const int c0 = blockIdx.x * 64, r0 = blockIdx.y * 64;
#pragma unroll
    for (int i = 0; i < 4; ++i) {
        int f = tid + i * 256;             // 1024 float4 = 64x64 tile
        int r = f >> 4, c4 = f & 15;
        const float4 v = *(const float4*)(s + (size_t)(r0 + r) * C + c0 + c4 * 4);
        __bf16* p = &t[r * 72 + c4 * 4];
        p[0] = f2bf(v.x); p[1] = f2bf(v.y); p[2] = f2bf(v.z); p[3] = f2bf(v.w);
    }
    __syncthreads();
#pragma unroll
    for (int i = 0; i < 4; ++i) {
        int f = tid + i * 256;
        int oc = f >> 4, g4 = f & 15;      // oc = src col, g4*4.. = src rows
        v4bf o;
        o[0] = t[(g4 * 4 + 0) * 72 + oc];
        o[1] = t[(g4 * 4 + 1) * 72 + oc];
        o[2] = t[(g4 * 4 + 2) * 72 + oc];
        o[3] = t[(g4 * 4 + 3) * 72 + oc];
        *(v4bf*)(d + (size_t)(c0 + oc) * R + r0 + g4 * 4) = o;
    }
}

// ---------------------------------------------------------------------------
// C = A * B with A (MxK row-major bf16) and Bt (NxK row-major bf16, i.e. B^T).
//   OUT_F32=1          : C f32 MxN row-major (final output)
//   OUT_F32=0,STORE_T=1: C bf16 stored TRANSPOSED (NxM) -> contiguous v8bf
//   OUT_F32=0,STORE_T=0: C bf16 MxN row-major
// M,N multiples of 128; K multiple of 64.
template <bool OUT_F32, bool STORE_T>
__global__ __launch_bounds__(256)
void gemm_bf16_nt(const __bf16* __restrict__ A,
                  const __bf16* __restrict__ Bt,
                  void* __restrict__ Cv,
                  int M, int N, int K)
{
    __shared__ __bf16 As[2][BM * SA];
    __shared__ __bf16 Bs[2][BN * SB];

    const int tid  = threadIdx.x;
    const int lane = tid & 31;
    const int wid  = tid >> 5;
    const int l16  = lane & 15;
    const bool hiHalf = (lane & 16) != 0;

    const int blockN = blockIdx.x * BN;
    const int blockM = blockIdx.y * BM;
    const int wmBase = (wid & 3) * 32;   // 4 waves tile M
    const int wnBase = (wid >> 2) * 64;  // 2 waves tile N

    v8f acc[2][4] = {};

    auto stage = [&](int k0, int b) {
#pragma unroll
        for (int i = 0; i < 4; ++i) {     // A tile: 128 x 64 halves = 1024 chunks
            int f = tid + i * 256;
            int row = f >> 3, c = f & 7;
            async_copy16(A + (size_t)(blockM + row) * K + k0 + c * 8,
                         &As[b][row * SA + c * 8]);
        }
#pragma unroll
        for (int i = 0; i < 4; ++i) {     // Bt tile: 128 x 64 halves
            int f = tid + i * 256;
            int row = f >> 3, c = f & 7;
            async_copy16(Bt + (size_t)(blockN + row) * K + k0 + c * 8,
                         &Bs[b][row * SB + c * 8]);
        }
    };

    const int T = K / BK;
    int buf = 0;
    stage(0, 0);

    for (int t = 0; t < T; ++t) {
        wait_async_all();
        __syncthreads();                  // tile t visible; buf^1 free
        if (t + 1 < T) stage((t + 1) * BK, buf ^ 1);

#pragma unroll
        for (int kk = 0; kk < 2; ++kk) {
            // A frag: lanes 0-15: M=lane, K {0..7,16..23}; lanes 16-31: K+8
            const int kA = kk * 32 + (hiHalf ? 8 : 0);
            v16bf afrag[2];
#pragma unroll
            for (int wm = 0; wm < 2; ++wm) {
                const __bf16* pa = &As[buf][(wmBase + wm * 16 + l16) * SA + kA];
                afrag[wm] = cat8(*(const v8bf*)pa, *(const v8bf*)(pa + 16));
            }
            // B frag: lanes 0-15: N=lane, K {0..7,16..23}; lanes 16-31: K+8
            const int kB = kk * 32 + (hiHalf ? 8 : 0);
#pragma unroll
            for (int wn = 0; wn < 4; ++wn) {
                const __bf16* pb = &Bs[buf][(wnBase + wn * 16 + l16) * SB + kB];
                v16bf bfrag = cat8(*(const v8bf*)pb, *(const v8bf*)(pb + 16));
#pragma unroll
                for (int wm = 0; wm < 2; ++wm) {
                    acc[wm][wn] = __builtin_amdgcn_wmma_f32_16x16x32_bf16(
                        false, afrag[wm], false, bfrag,
                        (short)0, acc[wm][wn], false, false);
                }
            }
        }
        buf ^= 1;
    }

    // C/D layout: VGPR r -> row r (lanes 0-15) / r+8 (lanes 16-31), lane -> col
#pragma unroll
    for (int wm = 0; wm < 2; ++wm) {
        const int rowb = blockM + wmBase + wm * 16 + (hiHalf ? 8 : 0);
#pragma unroll
        for (int wn = 0; wn < 4; ++wn) {
            const int col = blockN + wnBase + wn * 16 + l16;
            if constexpr (OUT_F32) {
                float* cp = (float*)Cv + (size_t)rowb * N + col;
#pragma unroll
                for (int r = 0; r < 8; ++r) cp[(size_t)r * N] = acc[wm][wn][r];
            } else if constexpr (STORE_T) {
                v8bf o;
#pragma unroll
                for (int r = 0; r < 8; ++r) o[r] = f2bf(acc[wm][wn][r]);
                *(v8bf*)((__bf16*)Cv + (size_t)col * M + rowb) = o;  // C^T, 16B
            } else {
                __bf16* cp = (__bf16*)Cv + (size_t)rowb * N + col;
#pragma unroll
                for (int r = 0; r < 8; ++r) cp[(size_t)r * N] = f2bf(acc[wm][wn][r]);
            }
        }
    }
}

// ---------------------------------------------------------------------------
extern "C" void kernel_launch(void* const* d_in, const int* in_sizes, int n_in,
                              void* d_out, int out_size, void* d_ws, size_t ws_size,
                              hipStream_t stream) {
    (void)in_sizes; (void)n_in; (void)out_size; (void)ws_size;

    const float* x  = (const float*)d_in[0];   // 8192 x 1024
    const float* WQ = (const float*)d_in[1];   // 1024 x 1024
    const float* WK = (const float*)d_in[2];
    const float* WV = (const float*)d_in[3];
    float* out = (float*)d_out;                // 8192 x 1024 f32

    constexpr int SEQ = 8192, D = 1024;
    constexpr size_t XY = (size_t)SEQ * D;     // 8,388,608
    constexpr size_t D2 = (size_t)D * D;       // 1,048,576

    __bf16* W    = (__bf16*)d_ws;
    __bf16* Xh   = W;                    // SEQ x D   (A of final GEMM)
    __bf16* XhT  = W + XY;               // D x SEQ   (A and B^T of G-GEMM)
    __bf16* Wqh  = W + 2 * XY;           // D x D     WQ row-major
    __bf16* WkT  = Wqh + D2;             // D x D     WK^T row-major
    __bf16* WvT  = WkT + D2;             // D x D     WV^T row-major
    __bf16* Gm   = WvT + D2;             // D x D     G = X^T X (row-major)
    __bf16* Pp   = Gm + D2;              // D x D     ping (T^T, then M^T)
    // S^T reuses Gm's slot after G is consumed.

    const dim3 blk(256);

    // ---- pre-pass: single bf16 conversion of every operand ----
    cvt_f32_bf16<<<dim3(XY / 2048), blk, 0, stream>>>(x,  Xh);
    cvt_f32_bf16<<<dim3(D2 / 2048), blk, 0, stream>>>(WQ, Wqh);
    tr_f32_bf16<<<dim3(D / 64, SEQ / 64), blk, 0, stream>>>(x,  XhT, SEQ, D);
    tr_f32_bf16<<<dim3(D / 64, D  / 64), blk, 0, stream>>>(WK, WkT, D, D);
    tr_f32_bf16<<<dim3(D / 64, D  / 64), blk, 0, stream>>>(WV, WvT, D, D);

    const dim3 gSq(D / BN, D / BM);            // 8 x 8
    const dim3 gOut(D / BN, SEQ / BM);         // 8 x 64

    // 1. G = X^T X          (A = XhT, B^T = XhT)            -> bf16 row-major
    gemm_bf16_nt<false, false><<<gSq, blk, 0, stream>>>(XhT, XhT, Gm, D, D, SEQ);
    // 2. T = G WV           (B^T = WvT)                     -> store T^T
    gemm_bf16_nt<false, true ><<<gSq, blk, 0, stream>>>(Gm, WvT, Pp, D, D, D);
    // 3. S = WK^T T         (A = WkT, B^T = T^T)            -> store S^T (over G)
    gemm_bf16_nt<false, true ><<<gSq, blk, 0, stream>>>(WkT, Pp, Gm, D, D, D);
    // 4. M = WQ S           (B^T = S^T)                     -> store M^T
    gemm_bf16_nt<false, true ><<<gSq, blk, 0, stream>>>(Wqh, Gm, Pp, D, D, D);
    // 5. out = X M          (A = Xh, B^T = M^T)             -> f32
    gemm_bf16_nt<true,  false><<<gOut, blk, 0, stream>>>(Xh, Pp, out, SEQ, D, D);
}
